// UGCG_GAT_50697793962354
// MI455X (gfx1250) — matrix operand order repaired and checked by reading, not compiled, not run on hardware
//
#include <hip/hip_runtime.h>

// ---------------------------------------------------------------------------
// Problem constants (from the reference)
// ---------------------------------------------------------------------------
constexpr int NN    = 50000;
constexpr int DEG   = 16;
constexpr int IN_F  = 512;
constexpr int HID   = 128;
constexpr int OUT_F = 64;
constexpr float SLOPE = 0.2f;

// ---------------------------------------------------------------------------
// Vector types for WMMA fragments (wave32, gfx1250)
// ---------------------------------------------------------------------------
typedef __attribute__((ext_vector_type(16))) __bf16 v16bf;
typedef __attribute__((ext_vector_type(8)))  __bf16 v8bf;
typedef __attribute__((ext_vector_type(8)))  float  v8f;

// ---------------------------------------------------------------------------
// Prep: convert W[K][Ncols] (f32 row-major) -> Wt[Ncols][K] (bf16), so a
// WMMA B-fragment (two runs of 8 consecutive K for one output column) is two contiguous
// 16-byte loads.
// ---------------------------------------------------------------------------
__global__ void convert_w_bf16_t(const float* __restrict__ W,
                                 __bf16* __restrict__ Wt, int K, int Ncols) {
    int i = blockIdx.x * 256 + threadIdx.x;
    if (i >= K * Ncols) return;
    int k = i / Ncols, n = i % Ncols;
    Wt[(size_t)n * K + k] = (__bf16)W[i];
}

// ---------------------------------------------------------------------------
// BF16 WMMA GEMM with fused attention projection:
//   H[M][NDIM] = X[M][KDIM] @ W ; el = H.attn_l ; er = H.attn_r
//
//   - 128-row tile per block, 256 threads = 8 waves, wave owns 16 rows x NDIM
//   - K stepped by 32 (one v_wmma_f32_16x16x32_bf16 depth)
//   - A slab staged through ping-pong LDS buffers (f32 -> bf16 on the fly);
//     the next slab's global loads are issued BEFORE the wmma chain and only
//     waited on at the ds_store, overlapping HBM/L2 latency with math
//   - All NT B-fragments are hoisted into registers before the wmma chain so
//     the loads form one clause with a single wait instead of load->wait->wmma
// ---------------------------------------------------------------------------
template <int KDIM, int NDIM>
__global__ __launch_bounds__(256)
void gemm_bf16_wmma(const float* __restrict__ X, const __bf16* __restrict__ Wt,
                    const float* __restrict__ al, const float* __restrict__ ar,
                    float* __restrict__ H, float* __restrict__ el,
                    float* __restrict__ er, int M) {
    constexpr int BM     = 128;
    constexpr int KSTEP  = 32;
    constexpr int NSTEP  = KDIM / KSTEP;
    constexpr int NT     = NDIM / 16;
    constexpr int LDSROW = 40;   // 32 bf16 + pad (80 B stride: 16B-aligned, conflict-free)

    __shared__ __bf16 sA[2][BM * LDSROW];

    const int tid      = threadIdx.x;
    const int wid      = tid >> 5;
    const int lane     = tid & 31;
    const int half     = lane >> 4;          // K-half select (ISA A layout)
    const int r        = lane & 15;          // row within 16x16 tile / column n
    const int waveRow  = wid * 16;
    const int blockRow = blockIdx.x * BM;

    // Staging geometry: thread covers rows {p*32 + tid/8}, cols (tid&7)*4..+3
    const int sRow = tid >> 3;
    const int sCol = (tid & 7) * 4;

    auto stage_load = [&](int k0, float4 (&pre)[4]) {
        #pragma unroll
        for (int p = 0; p < 4; ++p) {
            int grow = blockRow + p * 32 + sRow;
            pre[p] = make_float4(0.f, 0.f, 0.f, 0.f);
            if (grow < M)
                pre[p] = *(const float4*)(X + (size_t)grow * KDIM + k0 + sCol);
        }
    };
    auto stage_store = [&](int buf, const float4 (&pre)[4]) {
        #pragma unroll
        for (int p = 0; p < 4; ++p) {
            __bf16* dst = &sA[buf][(p * 32 + sRow) * LDSROW + sCol];
            dst[0] = (__bf16)pre[p].x; dst[1] = (__bf16)pre[p].y;
            dst[2] = (__bf16)pre[p].z; dst[3] = (__bf16)pre[p].w;
        }
    };

    v8f acc[NT] = {};
    float4 pre[4];

    stage_load(0, pre);
    stage_store(0, pre);

    for (int ks = 0; ks < NSTEP; ++ks) {
        const int cur = ks & 1;
        __syncthreads();                      // sA[cur] population visible

        // Issue next slab's global loads now; wait only at stage_store below.
        if (ks + 1 < NSTEP) stage_load((ks + 1) * KSTEP, pre);

        // A fragment, per ISA 16-bit A layout:
        //   lanes 0-15: K = {0..7} in v0-3, {16..23} in v4-7 (row = r)
        //   lanes 16-31: K = {8..15}, {24..31}
        const __bf16* arow = &sA[cur][(waveRow + r) * LDSROW];
        v8bf lo = *(const v8bf*)(arow + half * 8);
        v8bf hi = *(const v8bf*)(arow + 16 + half * 8);
        v16bf a;
        #pragma unroll
        for (int i = 0; i < 8; ++i) { a[i] = lo[i]; a[i + 8] = hi[i]; }

        // Hoist all B fragments (column n = t*16 + r, K = k0 + half*8 + {0..7}, k0 + 16 + half*8 + {0..7})
        v16bf bfr[NT];
        #pragma unroll
        for (int t = 0; t < NT; ++t) {
            const __bf16* bp =
                Wt + (size_t)(t * 16 + r) * KDIM + ks * KSTEP + half * 8;
            v8bf blo = *(const v8bf*)bp;
            v8bf bhi = *(const v8bf*)(bp + 16);
            #pragma unroll
            for (int i = 0; i < 8; ++i) { bfr[t][i] = blo[i]; bfr[t][i + 8] = bhi[i]; }
        }

        #pragma unroll
        for (int t = 0; t < NT; ++t)
            acc[t] = __builtin_amdgcn_wmma_f32_16x16x32_bf16(
                false, a, false, bfr[t], (short)0, acc[t], false, false);

        if (ks + 1 < NSTEP) stage_store(cur ^ 1, pre);
    }

    // Epilogue. C layout: element i of v8f is (M = half*8 + i, N = r).
    // Fused attention projection: el/er partial dot across the wave's columns.
    float pl[8], pr[8];
    #pragma unroll
    for (int i = 0; i < 8; ++i) { pl[i] = 0.f; pr[i] = 0.f; }

    #pragma unroll
    for (int t = 0; t < NT; ++t) {
        const float wl = al[t * 16 + r];
        const float wr = ar[t * 16 + r];
        #pragma unroll
        for (int i = 0; i < 8; ++i) {
            int row = blockRow + waveRow + half * 8 + i;
            float hv = acc[t][i];
            pl[i] += hv * wl;
            pr[i] += hv * wr;
            if (row < M) H[(size_t)row * NDIM + t * 16 + r] = hv;
        }
    }
    // Reduce over the 16 columns held by this half-wave (xor masks stay in group)
    #pragma unroll
    for (int m = 8; m >= 1; m >>= 1) {
        #pragma unroll
        for (int i = 0; i < 8; ++i) {
            pl[i] += __shfl_xor(pl[i], m, 32);
            pr[i] += __shfl_xor(pr[i], m, 32);
        }
    }
    if (r == 0) {
        #pragma unroll
        for (int i = 0; i < 8; ++i) {
            int row = blockRow + waveRow + half * 8 + i;
            if (row < M) { el[row] = pl[i]; er[row] = pr[i]; }
        }
    }
}

// ---------------------------------------------------------------------------
// Per-node edge softmax + weighted aggregation (one wave32 per node).
// Uniform degree 16: edges of v are [16v, 16v+16). Lanes 0-15 and 16-31
// redundantly hold edge j = lane&15 so all xor-shuffle reductions (masks<=8)
// stay inside 16-lane groups under wave32.
// ---------------------------------------------------------------------------
template <int F>
__global__ __launch_bounds__(256)
void gat_edge_agg(const int* __restrict__ col_idx, const float* __restrict__ H,
                  const float* __restrict__ el, const float* __restrict__ er,
                  const float* __restrict__ bias, float* __restrict__ out,
                  int n) {
    constexpr int FP = F / 32;
    int wid = threadIdx.x >> 5, lane = threadIdx.x & 31;
    int v = blockIdx.x * 8 + wid;
    if (v >= n) return;

    int j   = lane & 15;
    int src = col_idx[v * DEG + j];
    float s = el[v] + er[src];
    s = s > 0.f ? s : SLOPE * s;             // leaky_relu

    float mx = s;
    #pragma unroll
    for (int m = 8; m >= 1; m >>= 1) mx = fmaxf(mx, __shfl_xor(mx, m, 32));
    float ex  = __expf(s - mx);
    float sum = ex;
    #pragma unroll
    for (int m = 8; m >= 1; m >>= 1) sum += __shfl_xor(sum, m, 32);
    float alpha = ex / sum;

    float acc[FP];
    #pragma unroll
    for (int i = 0; i < FP; ++i) acc[i] = 0.f;

    #pragma unroll
    for (int jj = 0; jj < DEG; ++jj) {
        int   sj = __shfl(src,   jj, 32);
        float aj = __shfl(alpha, jj, 32);
        const float* hp = H + (size_t)sj * F + lane * FP;  // coalesced row gather (L2-resident)
        #pragma unroll
        for (int i = 0; i < FP; ++i) acc[i] += aj * hp[i];
    }

    #pragma unroll
    for (int i = 0; i < FP; ++i)
        out[(size_t)v * F + lane * FP + i] = acc[i] + bias[lane * FP + i];
}

// ---------------------------------------------------------------------------
// Workspace layout
// ---------------------------------------------------------------------------
constexpr size_t align256(size_t x) { return (x + 255) & ~size_t(255); }

constexpr size_t OFF_WT1 = 0;                                            // bf16 [HID][IN_F]
constexpr size_t OFF_WT2 = align256(OFF_WT1 + (size_t)IN_F * HID * 2);   // bf16 [OUT_F][HID]
constexpr size_t OFF_H1  = align256(OFF_WT2 + (size_t)HID * OUT_F * 2);  // f32 [NN][HID]
constexpr size_t OFF_X2  = align256(OFF_H1 + (size_t)NN * HID * 4);      // f32 [NN][HID]
constexpr size_t OFF_H2  = align256(OFF_X2 + (size_t)NN * HID * 4);      // f32 [NN][OUT_F]
constexpr size_t OFF_EL1 = align256(OFF_H2 + (size_t)NN * OUT_F * 4);
constexpr size_t OFF_ER1 = align256(OFF_EL1 + (size_t)NN * 4);
constexpr size_t OFF_EL2 = align256(OFF_ER1 + (size_t)NN * 4);
constexpr size_t OFF_ER2 = align256(OFF_EL2 + (size_t)NN * 4);

extern "C" void kernel_launch(void* const* d_in, const int* in_sizes, int n_in,
                              void* d_out, int out_size, void* d_ws,
                              size_t ws_size, hipStream_t stream) {
    (void)in_sizes; (void)n_in; (void)out_size; (void)ws_size;

    // setup_inputs() order:
    // 0 row_ptr (unused: uniform degree), 1 col_idx, 2 input_feature,
    // 3 W1, 4 attn_l1, 5 attn_r1, 6 bias1, 7 W2, 8 attn_l2, 9 attn_r2, 10 bias2
    const int*   col = (const int*)d_in[1];
    const float* x   = (const float*)d_in[2];
    const float* W1  = (const float*)d_in[3];
    const float* al1 = (const float*)d_in[4];
    const float* ar1 = (const float*)d_in[5];
    const float* b1  = (const float*)d_in[6];
    const float* W2  = (const float*)d_in[7];
    const float* al2 = (const float*)d_in[8];
    const float* ar2 = (const float*)d_in[9];
    const float* b2  = (const float*)d_in[10];
    float* out = (float*)d_out;

    char* ws = (char*)d_ws;
    __bf16* Wt1 = (__bf16*)(ws + OFF_WT1);
    __bf16* Wt2 = (__bf16*)(ws + OFF_WT2);
    float*  H1  = (float*)(ws + OFF_H1);
    float*  X2  = (float*)(ws + OFF_X2);
    float*  H2  = (float*)(ws + OFF_H2);
    float*  el1 = (float*)(ws + OFF_EL1);
    float*  er1 = (float*)(ws + OFF_ER1);
    float*  el2 = (float*)(ws + OFF_EL2);
    float*  er2 = (float*)(ws + OFF_ER2);

    const dim3 blk(256);
    const dim3 gemmGrid((NN + 127) / 128);
    const dim3 nodeGrid((NN + 7) / 8);

    // Layer 1
    convert_w_bf16_t<<<(IN_F * HID + 255) / 256, blk, 0, stream>>>(W1, Wt1, IN_F, HID);
    gemm_bf16_wmma<IN_F, HID><<<gemmGrid, blk, 0, stream>>>(x, Wt1, al1, ar1,
                                                            H1, el1, er1, NN);
    gat_edge_agg<HID><<<nodeGrid, blk, 0, stream>>>(col, H1, el1, er1, b1, X2, NN);

    // Layer 2
    convert_w_bf16_t<<<(HID * OUT_F + 255) / 256, blk, 0, stream>>>(W2, Wt2, HID, OUT_F);
    gemm_bf16_wmma<HID, OUT_F><<<gemmGrid, blk, 0, stream>>>(X2, Wt2, al2, ar2,
                                                             H2, el2, er2, NN);
    gat_edge_agg<OUT_F><<<nodeGrid, blk, 0, stream>>>(col, H2, el2, er2, b2, out, NN);
}
